// TimeDecoder_34763465294429
// MI455X (gfx1250) — matrix operand order repaired and checked by reading, not compiled, not run on hardware
//
#include <hip/hip_runtime.h>
#include <stdint.h>

typedef __attribute__((ext_vector_type(16))) __bf16 v16bf;
typedef __attribute__((ext_vector_type(8)))  float  v8f;

#define TN    4000
#define MDIM  1500
#define NDIM  1024
#define KPAD  1536
#define LROW  1544          // padded LDS row stride (ushorts) -> 3088 B, bank-shift 4
#define LLEN  2048512       // (T-1)*512 + 1024

__device__ __forceinline__ unsigned short f2bf(float f) {
    uint32_t u = __builtin_bit_cast(uint32_t, f);
    u += 0x7fffu + ((u >> 16) & 1u);          // round-to-nearest-even
    return (unsigned short)(u >> 16);
}

__global__ __launch_bounds__(256) void zero_f4(float4* __restrict__ out, int n4) {
    int i = blockIdx.x * 256 + threadIdx.x;
    if (i < n4) out[i] = make_float4(0.f, 0.f, 0.f, 0.f);
}

// W [1024,1500] f32 -> Wb [1024,1536] bf16 (zero padded K tail)
__global__ __launch_bounds__(256) void cvt_w(const float* __restrict__ W,
                                             unsigned short* __restrict__ Wb) {
    int k = blockIdx.x * 256 + threadIdx.x;   // 0..1535
    int n = blockIdx.y;                        // 0..1023
    unsigned short v = 0;
    if (k < MDIM) v = f2bf(W[(size_t)n * MDIM + k]);
    Wb[(size_t)n * KPAD + k] = v;
}

struct FragQ { uint4 q0, q1; };               // 32 B = one 16-value bf16 fragment

__device__ __forceinline__ v16bf fv(const FragQ& f) {
    return __builtin_bit_cast(v16bf, f);
}

__device__ __forceinline__ void load_bf(FragQ bfr[2], const unsigned short* w0,
                                        const unsigned short* w1, int kk) {
    const uint4* p0 = (const uint4*)(w0 + kk);
    const uint4* p1 = (const uint4*)(w1 + kk);
    bfr[0].q0 = p0[0]; bfr[0].q1 = p0[1];
    bfr[1].q0 = p1[0]; bfr[1].q1 = p1[1];
}

__device__ __forceinline__ void load_af(FragQ afr[4],
                                        const unsigned short* const arow[4], int kk) {
    #pragma unroll
    for (int i = 0; i < 4; ++i) {
        afr[i].q0 = *(const uint4*)(arow[i] + kk);
        afr[i].q1 = *(const uint4*)(arow[i] + kk + 8);
    }
}

__device__ __forceinline__ void mma8(v8f acc[2][4], const FragQ afr[4], const FragQ bfr[2]) {
    #pragma unroll
    for (int i = 0; i < 4; ++i) {
        #pragma unroll
        for (int s = 0; s < 2; ++s) {
            acc[s][i] = __builtin_amdgcn_wmma_f32_16x16x32_bf16(
                false, fv(afr[i]), false, fv(bfr[s]), (short)0, acc[s][i], false, false);
        }
    }
}

__global__ __launch_bounds__(256) void gemm_oadd(
    const float* __restrict__ A,              // [32000,1500]
    const float* __restrict__ norm,           // [32000]
    const unsigned short* __restrict__ Wb,    // [1024,1536] bf16
    const float* __restrict__ bias,           // [1024]
    float* __restrict__ out)                  // [8, 2048512]
{
    __shared__ alignas(16) unsigned short sA[64 * LROW];   // 193 KB (CDNA5: 320 KB/WGP)
    const int tid = threadIdx.x;
    const int r0  = blockIdx.x * 64;

    // ---- Stage A tile: 64 rows x 1500 f32 -> bf16 in LDS, pad K to 1536 with 0 ----
    {
        const float4* Ag = (const float4*)(A + (size_t)r0 * MDIM);  // row stride 375 float4
        for (int idx = tid; idx < 64 * 375; idx += 256) {
            int row = idx / 375, c = idx - row * 375;
            float4 f = Ag[(size_t)row * 375 + c];
            uint32_t lo = (uint32_t)f2bf(f.x) | ((uint32_t)f2bf(f.y) << 16);
            uint32_t hi = (uint32_t)f2bf(f.z) | ((uint32_t)f2bf(f.w) << 16);
            *(uint2*)&sA[row * LROW + c * 4] = make_uint2(lo, hi);
        }
        for (int idx = tid; idx < 64 * 44; idx += 256) {   // zero K = 1500..1543
            int row = idx / 44, c = 1500 + (idx - row * 44);
            sA[row * LROW + c] = 0;
        }
    }
    __syncthreads();

    const int lane = tid & 31;
    const int wav  = tid >> 5;        // 0..7 -> 16-col strip pair within the 256-col tile
    const int lhi  = lane >> 4;       // K-half select: this code's A/B order k = 16*lhi + s
    const int ln   = lane & 15;

    // Loop-invariant LDS row-tile bases: every ds_load becomes base-reg + imm16 offset.
    const unsigned short* arow[4];
    #pragma unroll
    for (int i = 0; i < 4; ++i)
        arow[i] = &sA[(i * 16 + ln) * LROW + lhi * 16];

    for (int ct = 0; ct < 4; ++ct) {
        const int nbase = ct * 256 + wav * 16 + ln;            // strip 0 column
        const unsigned short* wrow0 = Wb + (size_t)nbase * KPAD + lhi * 16;
        const unsigned short* wrow1 = wrow0 + (size_t)128 * KPAD;  // strip 1 (+128 cols)

        v8f acc[2][4];
        #pragma unroll
        for (int s = 0; s < 2; ++s)
            #pragma unroll
            for (int i = 0; i < 4; ++i)
                acc[s][i] = (v8f){0.f,0.f,0.f,0.f,0.f,0.f,0.f,0.f};

        // ---- Software-pipelined K loop: double-buffered fragments ----
        FragQ a0[4], a1[4], b0[2], b1[2];
        load_bf(b0, wrow0, wrow1, 0);
        load_af(a0, arow, 0);
        int kk = 0;
        for (; kk + 64 < KPAD; kk += 64) {
            load_bf(b1, wrow0, wrow1, kk + 32);
            load_af(a1, arow, kk + 32);
            mma8(acc, a0, b0);                 // consume buf0 while buf1 in flight
            load_bf(b0, wrow0, wrow1, kk + 64);
            load_af(a0, arow, kk + 64);
            mma8(acc, a1, b1);                 // consume buf1 while buf0 in flight
        }
        load_bf(b1, wrow0, wrow1, kk + 32);    // kk = 1472 here; load 1504
        load_af(a1, arow, kk + 32);
        mma8(acc, a0, b0);                     // k = 1472
        mma8(acc, a1, b1);                     // k = 1504

        // ---- Fused epilogue: (+bias) * norm * edge_scale, overlap-add via f32 atomics ----
        #pragma unroll
        for (int s = 0; s < 2; ++s) {
            const int ncol = nbase + s * 128;
            const float bval = bias[ncol];
            #pragma unroll
            for (int i = 0; i < 4; ++i) {
                #pragma unroll
                for (int r = 0; r < 8; ++r) {
                    int row = r0 + i * 16 + r + lhi * 8;  // C: VGPR r -> M = r + 8*(lane>=16)
                    int b = row / TN;
                    int t = row - b * TN;
                    float sc = ((t == 0 && ncol < 512) || (t == TN - 1 && ncol >= 512))
                               ? 1.0f : 0.5f;
                    float val = (acc[s][i][r] + bval) * norm[row] * sc;
                    float* p = out + (size_t)b * LLEN + (size_t)t * 512 + ncol;
                    asm volatile("global_atomic_add_f32 %0, %1, off"
                                 :: "v"(p), "v"(val) : "memory");
                }
            }
        }
    }
}

extern "C" void kernel_launch(void* const* d_in, const int* in_sizes, int n_in,
                              void* d_out, int out_size, void* d_ws, size_t ws_size,
                              hipStream_t stream) {
    (void)in_sizes; (void)n_in; (void)ws_size;
    const float* A    = (const float*)d_in[0];   // waveform_encoding [8,4000,1500]
    const float* norm = (const float*)d_in[1];   // waveform_norm     [8,4000,1]
    const float* W    = (const float*)d_in[2];   // W [1024,1500]
    const float* bias = (const float*)d_in[3];   // b [1024]
    float* out = (float*)d_out;
    unsigned short* Wb = (unsigned short*)d_ws;  // 1024*1536*2 = 3 MB scratch

    int n4 = out_size / 4;                       // 16,388,096 / 4 (exact)
    zero_f4<<<(n4 + 255) / 256, 256, 0, stream>>>((float4*)out, n4);

    dim3 gw(KPAD / 256, NDIM);
    cvt_w<<<gw, 256, 0, stream>>>(W, Wb);

    gemm_oadd<<<500, 256, 0, stream>>>(A, norm, Wb, bias, out);
}
